// TGNNodeClassifier_86577950753301
// MI455X (gfx1250) — matrix operand order repaired and checked by reading, not compiled, not run on hardware
//
#include <hip/hip_runtime.h>
#include <hip/hip_bf16.h>
#include <math.h>

// ---------------- model constants ----------------
#define NNODES 50000
#define NEDGES 8192
#define INDIM  256
#define HIDDIM 256
#define TDIM   32
#define NCLS   64
#define ATTN_SCALE 0.0625f   // 1/sqrt(256)

typedef __attribute__((ext_vector_type(16))) _Float16 v16h;
typedef __attribute__((ext_vector_type(8)))  _Float16 v8h;
typedef __attribute__((ext_vector_type(8)))  float    v8f;

union V16U { v16h v; v8h h[2]; };

// LDS tile geometry (halfwords). Rows padded 32 -> 40 halfwords (80B stride =
// 20-bank rotation; the 16 rows a b128 ds read touches hit distinct bank groups).
#define ROWH   40
#define ABUF_H (128 * ROWH)          // A buffer: 5120 H = 10240 B
#define BOFF_H (2 * ABUF_H)          // B region starts at 20480 B
#define BBUF_H (64 * ROWH)           // B buffer: 2560 H = 5120 B
#define SMEM_H (2 * ABUF_H + 2 * BBUF_H)   // 15360 H = 30720 B

// ---------------- small utility kernels ----------------
__global__ void cvt_f16_kernel(const float* __restrict__ in, _Float16* __restrict__ out, long n) {
    long i = (long)blockIdx.x * blockDim.x + threadIdx.x;
    if (i < n) out[i] = (_Float16)in[i];
}

__global__ void zero_u32_kernel(unsigned* __restrict__ p, long n) {
    long i = (long)blockIdx.x * blockDim.x + threadIdx.x;
    if (i < n) p[i] = 0u;
}

// ---------------- bitonic sort of packed (dst,time,edge) keys ----------------
// key = dst[e] << 45 | orderedbits(time[e]) << 13 | e
// times are uniform in [0,1): positive floats -> raw bits are order-preserving.
__global__ void sort_edges_kernel(const float* __restrict__ et,
                                  const int* __restrict__ dstA,
                                  unsigned long long* __restrict__ keysOut) {
    __shared__ unsigned long long sk[NEDGES];   // 64 KB LDS
    const int t = threadIdx.x;                  // 1024 threads
    for (int i = t; i < NEDGES; i += 1024) {
        unsigned tb = __float_as_uint(et[i]);
        unsigned long long key = ((unsigned long long)(unsigned)dstA[i] << 45)
                               | ((unsigned long long)tb << 13)
                               | (unsigned long long)(unsigned)i;
        sk[i] = key;
    }
    __syncthreads();
    for (unsigned ksz = 2; ksz <= NEDGES; ksz <<= 1) {
        for (unsigned j = ksz >> 1; j > 0; j >>= 1) {
            for (int i = t; i < NEDGES; i += 1024) {
                int ixj = i ^ (int)j;
                if (ixj > i) {
                    bool asc = ((i & ksz) == 0);
                    unsigned long long a = sk[i], b = sk[ixj];
                    if ((a > b) == asc) { sk[i] = b; sk[ixj] = a; }
                }
            }
            __syncthreads();
        }
    }
    for (int i = t; i < NEDGES; i += 1024) keysOut[i] = sk[i];
}

__global__ void chain_starts_kernel(const unsigned long long* __restrict__ keys,
                                    int* __restrict__ starts, int* __restrict__ counter) {
    int p = blockIdx.x * blockDim.x + threadIdx.x;
    if (p >= NEDGES) return;
    unsigned d = (unsigned)(keys[p] >> 45);
    bool st = (p == 0) || ((unsigned)(keys[p - 1] >> 45) != d);
    if (st) {
        int idx = atomicAdd(counter, 1);
        starts[idx] = p;
    }
}

// ---------------- build msgs = [x[src] , relu(t*Wt+bt)] in f16 ----------------
__global__ void time_msgs_kernel(const _Float16* __restrict__ xh,
                                 const int* __restrict__ srcA,
                                 const float* __restrict__ et,
                                 const float* __restrict__ timeW,
                                 const float* __restrict__ timeB,
                                 _Float16* __restrict__ msgs) {
    int e = blockIdx.x, t = threadIdx.x;           // 288 threads
    long s = srcA[e];
    if (t < INDIM) {
        msgs[(long)e * (INDIM + TDIM) + t] = xh[s * INDIM + t];
    } else {
        int tt = t - INDIM;
        float v = et[e] * timeW[tt] + timeB[tt];
        v = fmaxf(v, 0.f);
        msgs[(long)e * (INDIM + TDIM) + t] = (_Float16)v;
    }
}

// ---------------- WMMA GEMM: C[M,N] = A[M,K] @ W[N,K]^T + bias ----------------
// 256 threads = 8 waves; block tile 128x64; wave tile 16x64; K-step 32.
// A and B tiles both double-buffered in LDS via gfx1250 async-to-LDS; the
// K-loop has no global loads at all: ds_load_b128 fragment reads -> v_wmma.
// LDS address operands are DERIVED FROM the shared pointer (escapes into the
// asm -> compiler must honor the async writes; low 32 bits of a generic
// shared-space address are the LDS byte offset per ISA 10.2).

__device__ __forceinline__ void stage_async(const _Float16* __restrict__ A,
                                            const _Float16* __restrict__ W,
                                            _Float16* smem, int M,
                                            int m0, int n0, int k0, int K, int buf, int tid) {
    const int row   = tid >> 2;     // 0..63
    const int chunk = tid & 3;      // 0..3 (16B each)
    // A tile: 128 rows x 64B, coalesced; clamp edge rows (no OOB reads)
#pragma unroll
    for (int rr = 0; rr < 2; ++rr) {
        int r = row + rr * 64;
        long gr = m0 + r;
        if (gr >= M) gr = M - 1;
        unsigned long long ga =
            (unsigned long long)(size_t)(A + gr * (long)K + k0) + (unsigned long long)(chunk * 16);
        unsigned lo = (unsigned)(size_t)(smem + buf * ABUF_H + r * ROWH + chunk * 8);
        asm volatile("global_load_async_to_lds_b128 %0, %1, off"
                     :: "v"(lo), "v"(ga) : "memory");
    }
    // B tile: 64 weight rows x 64B (N is always a multiple of 64 -> in range)
    {
        unsigned long long ga =
            (unsigned long long)(size_t)(W + (long)(n0 + row) * K + k0) + (unsigned long long)(chunk * 16);
        unsigned lo = (unsigned)(size_t)(smem + BOFF_H + buf * BBUF_H + row * ROWH + chunk * 8);
        asm volatile("global_load_async_to_lds_b128 %0, %1, off"
                     :: "v"(lo), "v"(ga) : "memory");
    }
}

__device__ __forceinline__ void wait_async() {
    asm volatile("s_wait_asynccnt 0x0" ::: "memory");
}

__global__ void gemm_wmma_kernel(const _Float16* __restrict__ A,
                                 const _Float16* __restrict__ W,
                                 const float* __restrict__ bias,
                                 float* __restrict__ Cf,
                                 _Float16* __restrict__ Ch,
                                 int M, int K, int ldc, int colOff, int act) {
    __shared__ _Float16 smem[SMEM_H];        // 30 KB: A ping/pong + B ping/pong
    const int tid  = threadIdx.x;
    const int wid  = tid >> 5;
    const int lane = tid & 31;
    const int row  = lane & 15;
    const int half = lane >> 4;
    const int m0 = blockIdx.x * 128;
    const int n0 = blockIdx.y * 64;
    const int mw = m0 + wid * 16;            // this wave's M-tile base

    // prologue: stage first A+B tiles
    stage_async(A, W, smem, M, m0, n0, 0, K, 0, tid);
    wait_async();
    __syncthreads();

    v8f acc[4] = {};
    int buf = 0;
    for (int k0 = 0; k0 < K; k0 += 32) {
        const bool more = (k0 + 32) < K;
        if (more) stage_async(A, W, smem, M, m0, n0, k0 + 32, K, buf ^ 1, tid);  // overlaps WMMAs

        // A fragment (ds_load_b128 x2): row = mw+row, K slots half*16 + (0..15) contiguous
        V16U a;
        const _Float16* la = smem + buf * ABUF_H + (wid * 16 + row) * ROWH;
        a.h[0] = *(const v8h*)(la + half * 16);
        a.h[1] = *(const v8h*)(la + half * 16 + 8);
#pragma unroll
        for (int j = 0; j < 4; ++j) {
            // B fragment: column n0+j*16+row, K slots half*16 + (0..15) contiguous
            const _Float16* lb = smem + BOFF_H + buf * BBUF_H + (j * 16 + row) * ROWH + half * 16;
            V16U b;
            b.h[0] = *(const v8h*)(lb);
            b.h[1] = *(const v8h*)(lb + 8);
            acc[j] = __builtin_amdgcn_wmma_f32_16x16x32_f16(
                false, a.v, false, b.v, (short)0, acc[j], false, false);
        }
        if (more) wait_async();
        __syncthreads();
        buf ^= 1;
    }

#pragma unroll
    for (int j = 0; j < 4; ++j) {
        int n = n0 + j * 16 + row;
        float bs = bias ? bias[n] : 0.f;
#pragma unroll
        for (int v = 0; v < 8; ++v) {
            int m = mw + v + 8 * half;       // C/D layout: M = vgpr + 8*(lane/16)
            if (m < M) {
                float val = acc[j][v] + bs;
                if (act) val = fmaxf(val, 0.f);
                long idx = (long)m * ldc + colOff + n;
                if (Ch) Ch[idx] = (_Float16)val;
                else    Cf[idx] = val;
            }
        }
    }
}

// ---------------- GRU over per-dst chains (time order) ----------------
__global__ void gru_chains_kernel(const unsigned long long* __restrict__ keys,
                                  const int* __restrict__ starts,
                                  const int* __restrict__ counter,
                                  const float* __restrict__ gi,
                                  const float* __restrict__ Whh,
                                  const float* __restrict__ bhh,
                                  float* __restrict__ mem) {
    __shared__ float h[HIDDIM];
    if (blockIdx.x >= counter[0]) return;
    int p = starts[blockIdx.x];
    int t = threadIdx.x;                         // 256 threads, one per hidden dim
    int d = (int)(keys[p] >> 45);
    h[t] = mem[(long)d * HIDDIM + t];
    __syncthreads();
    while (p < NEDGES) {
        unsigned long long key = keys[p];
        if ((int)(key >> 45) != d) break;
        int e = (int)(key & 0x1FFFULL);
        const float* gie = gi + (long)e * (3 * HIDDIM);
        const float* w0 = Whh + (long)t * HIDDIM;
        const float* w1 = Whh + (long)(HIDDIM + t) * HIDDIM;
        const float* w2 = Whh + (long)(2 * HIDDIM + t) * HIDDIM;
        float s0 = 0.f, s1 = 0.f, s2 = 0.f;
#pragma unroll 4
        for (int k = 0; k < HIDDIM; ++k) {
            float hv = h[k];
            s0 += w0[k] * hv; s1 += w1[k] * hv; s2 += w2[k] * hv;
        }
        float rr = 1.f / (1.f + __expf(-(gie[t] + s0 + bhh[t])));
        float zz = 1.f / (1.f + __expf(-(gie[HIDDIM + t] + s1 + bhh[HIDDIM + t])));
        float nn = tanhf(gie[2 * HIDDIM + t] + rr * (s2 + bhh[2 * HIDDIM + t]));
        float hnew = (1.f - zz) * nn + zz * h[t];
        __syncthreads();
        h[t] = hnew;
        __syncthreads();
        ++p;
    }
    mem[(long)d * HIDDIM + t] = h[t];
}

__global__ void mem_to_emb_kernel(const float* __restrict__ mem, _Float16* __restrict__ emb0) {
    int n = blockIdx.x, c = threadIdx.x;
    emb0[(long)n * 512 + 256 + c] = (_Float16)mem[(long)n * 256 + c];
}

// ---------------- gather edge rows ----------------
__global__ void gather_rows_kernel(const _Float16* __restrict__ src,
                                   const int* __restrict__ idx,
                                   _Float16* __restrict__ out, int width) {
    int e = blockIdx.x;
    long r = idx[e];
    for (int c = threadIdx.x; c < width; c += blockDim.x)
        out[(long)e * width + c] = src[r * width + c];
}

// ---------------- segment softmax attention ----------------
__device__ __forceinline__ unsigned enc_f(float f) {
    unsigned i = __float_as_uint(f);
    return (i & 0x80000000u) ? ~i : (i | 0x80000000u);
}
__device__ __forceinline__ float dec_f(unsigned u) {
    unsigned i = (u & 0x80000000u) ? (u & 0x7FFFFFFFu) : ~u;
    return __uint_as_float(i);
}

__global__ void attn_alpha_kernel(const float* __restrict__ qe,
                                  const float* __restrict__ ke,
                                  const int* __restrict__ dstA,
                                  float* __restrict__ alpha,
                                  unsigned* __restrict__ menc) {
    int gw = (blockIdx.x * blockDim.x + threadIdx.x) >> 5;
    int lane = threadIdx.x & 31;
    if (gw >= NEDGES * 2) return;
    int e = gw >> 1, h = gw & 1;
    const float* qp = qe + (long)e * 512 + h * 256;
    const float* kp = ke + (long)e * 512 + h * 256;
    float s = 0.f;
    for (int c = lane; c < 256; c += 32) s += qp[c] * kp[c];
    for (int off = 16; off > 0; off >>= 1) s += __shfl_down(s, off, 32);
    if (lane == 0) {
        float al = s * ATTN_SCALE;
        alpha[e * 2 + h] = al;
        atomicMax(&menc[(long)dstA[e] * 2 + h], enc_f(al));
    }
}

__global__ void attn_exp_kernel(const float* __restrict__ alpha,
                                const int* __restrict__ dstA,
                                const unsigned* __restrict__ menc,
                                float* __restrict__ exv,
                                float* __restrict__ denom) {
    int i = blockIdx.x * blockDim.x + threadIdx.x;
    if (i >= NEDGES * 2) return;
    int e = i >> 1, h = i & 1;
    long d = dstA[e];
    float m = dec_f(menc[d * 2 + h]);
    float ex = __expf(alpha[i] - m);
    exv[i] = ex;
    atomicAdd(&denom[d * 2 + h], ex);
}

__global__ void attn_agg_kernel(const float* __restrict__ exv,
                                const float* __restrict__ denom,
                                const float* __restrict__ ve,
                                const int* __restrict__ dstA,
                                float* __restrict__ aggm) {
    int e = blockIdx.x, c = threadIdx.x;           // 256 threads
    long d = dstA[e];
    float w0 = exv[e * 2 + 0] / denom[d * 2 + 0];
    float w1 = exv[e * 2 + 1] / denom[d * 2 + 1];
    float val = 0.5f * (w0 * ve[(long)e * 512 + c] + w1 * ve[(long)e * 512 + 256 + c]);
    atomicAdd(&aggm[d * 256 + c], val);
}

__global__ void combine_kernel(const float* __restrict__ aggm,
                               const float* __restrict__ skip,
                               _Float16* __restrict__ out) {
    long i = (long)blockIdx.x * 256 + threadIdx.x;
    float v = aggm[i] + skip[i];
    out[i] = (_Float16)fmaxf(v, 0.f);
}

// ---------------- host side ----------------
static inline void* wsa(char* base, size_t& off, size_t bytes) {
    void* p = base + off;
    off = (off + bytes + 255) & ~(size_t)255;
    return p;
}

static inline void run_gemm(const _Float16* A, const _Float16* W, const float* bias,
                            float* Cf, _Float16* Ch, int M, int N, int K,
                            int ldc, int colOff, int act, hipStream_t s) {
    dim3 g((M + 127) / 128, N / 64);
    gemm_wmma_kernel<<<g, 256, 0, s>>>(A, W, bias, Cf, Ch, M, K, ldc, colOff, act);
}

static inline void run_cvt(const float* in, _Float16* out, long n, hipStream_t s) {
    cvt_f16_kernel<<<(unsigned)((n + 255) / 256), 256, 0, s>>>(in, out, n);
}

static inline void run_zero(void* p, long nDw, hipStream_t s) {
    zero_u32_kernel<<<(unsigned)((nDw + 255) / 256), 256, 0, s>>>((unsigned*)p, nDw);
}

extern "C" void kernel_launch(void* const* d_in, const int* in_sizes, int n_in,
                              void* d_out, int out_size, void* d_ws, size_t ws_size,
                              hipStream_t stream) {
    (void)in_sizes; (void)n_in; (void)out_size; (void)ws_size;
    const float* x        = (const float*)d_in[0];
    const int*   ei       = (const int*)d_in[1];
    const float* et       = (const float*)d_in[2];
    const float* memory   = (const float*)d_in[3];
    const float* gru_Whh  = (const float*)d_in[5];
    const float* gru_bih  = (const float*)d_in[6];
    const float* gru_bhh  = (const float*)d_in[7];
    const float* time_W   = (const float*)d_in[8];
    const float* time_b   = (const float*)d_in[9];
    const float* feat_b   = (const float*)d_in[11];
    const float* c0_qb = (const float*)d_in[13];
    const float* c0_kb = (const float*)d_in[15];
    const float* c0_vb = (const float*)d_in[17];
    const float* c0_sb = (const float*)d_in[19];
    const float* c1_qb = (const float*)d_in[21];
    const float* c1_kb = (const float*)d_in[23];
    const float* c1_vb = (const float*)d_in[25];
    const float* c1_sb = (const float*)d_in[27];
    const float* cls_b = (const float*)d_in[29];

    const int* srcA = ei;
    const int* dstA = ei + NEDGES;

    // workspace layout
    char* base = (char*)d_ws;
    size_t off = 0;
    _Float16* xh      = (_Float16*)wsa(base, off, (size_t)NNODES * INDIM * 2);
    _Float16* wih16   = (_Float16*)wsa(base, off, (size_t)768 * 288 * 2);
    _Float16* featw16 = (_Float16*)wsa(base, off, (size_t)256 * 256 * 2);
    _Float16* q0w16   = (_Float16*)wsa(base, off, (size_t)512 * 512 * 2);
    _Float16* k0w16   = (_Float16*)wsa(base, off, (size_t)512 * 512 * 2);
    _Float16* v0w16   = (_Float16*)wsa(base, off, (size_t)512 * 512 * 2);
    _Float16* s0w16   = (_Float16*)wsa(base, off, (size_t)256 * 512 * 2);
    _Float16* q1w16   = (_Float16*)wsa(base, off, (size_t)512 * 256 * 2);
    _Float16* k1w16   = (_Float16*)wsa(base, off, (size_t)512 * 256 * 2);
    _Float16* v1w16   = (_Float16*)wsa(base, off, (size_t)512 * 256 * 2);
    _Float16* s1w16   = (_Float16*)wsa(base, off, (size_t)256 * 256 * 2);
    _Float16* clsw16  = (_Float16*)wsa(base, off, (size_t)64 * 256 * 2);
    _Float16* msgs    = (_Float16*)wsa(base, off, (size_t)NEDGES * 288 * 2);
    float*    gi      = (float*)wsa(base, off, (size_t)NEDGES * 768 * 4);
    unsigned long long* keys = (unsigned long long*)wsa(base, off, (size_t)NEDGES * 8);
    int*      starts  = (int*)wsa(base, off, (size_t)NEDGES * 4);
    int*      counter = (int*)wsa(base, off, 256);
    float*    mem     = (float*)wsa(base, off, (size_t)NNODES * 256 * 4);
    _Float16* emb0    = (_Float16*)wsa(base, off, (size_t)NNODES * 512 * 2);
    float*    skip    = (float*)wsa(base, off, (size_t)NNODES * 256 * 4);
    _Float16* Xgd     = (_Float16*)wsa(base, off, (size_t)NEDGES * 512 * 2);
    _Float16* Xgs     = (_Float16*)wsa(base, off, (size_t)NEDGES * 512 * 2);
    float*    qe      = (float*)wsa(base, off, (size_t)NEDGES * 512 * 4);
    float*    ke      = (float*)wsa(base, off, (size_t)NEDGES * 512 * 4);
    float*    ve      = (float*)wsa(base, off, (size_t)NEDGES * 512 * 4);
    float*    alpha   = (float*)wsa(base, off, (size_t)NEDGES * 2 * 4);
    float*    exv     = (float*)wsa(base, off, (size_t)NEDGES * 2 * 4);
    unsigned* menc    = (unsigned*)wsa(base, off, (size_t)NNODES * 2 * 4);
    float*    denom   = (float*)wsa(base, off, (size_t)NNODES * 2 * 4);
    float*    aggm    = (float*)wsa(base, off, (size_t)NNODES * 256 * 4);
    _Float16* emb1    = (_Float16*)wsa(base, off, (size_t)NNODES * 256 * 2);
    _Float16* emb2    = (_Float16*)wsa(base, off, (size_t)NNODES * 256 * 2);

    // 1. precision casts (f32 -> f16) for WMMA operands
    run_cvt(x,                      xh,      (long)NNODES * INDIM, stream);
    run_cvt((const float*)d_in[4],  wih16,   768L * 288, stream);
    run_cvt((const float*)d_in[10], featw16, 256L * 256, stream);
    run_cvt((const float*)d_in[12], q0w16, 512L * 512, stream);
    run_cvt((const float*)d_in[14], k0w16, 512L * 512, stream);
    run_cvt((const float*)d_in[16], v0w16, 512L * 512, stream);
    run_cvt((const float*)d_in[18], s0w16, 256L * 512, stream);
    run_cvt((const float*)d_in[20], q1w16, 512L * 256, stream);
    run_cvt((const float*)d_in[22], k1w16, 512L * 256, stream);
    run_cvt((const float*)d_in[24], v1w16, 512L * 256, stream);
    run_cvt((const float*)d_in[26], s1w16, 256L * 256, stream);
    run_cvt((const float*)d_in[28], clsw16, 64L * 256, stream);

    // 2. memory buffer (input is the initial TGN memory state)
    hipMemcpyAsync(mem, memory, (size_t)NNODES * 256 * 4, hipMemcpyDeviceToDevice, stream);

    // 3. sort edges by (dst, time) — packs chain grouping + time order into one key
    sort_edges_kernel<<<1, 1024, 0, stream>>>(et, dstA, keys);
    run_zero(counter, 1, stream);
    chain_starts_kernel<<<(NEDGES + 255) / 256, 256, 0, stream>>>(keys, starts, counter);

    // 4. messages + batched input-half of GRU: gi = msgs @ Wih^T + bih  (WMMA)
    time_msgs_kernel<<<NEDGES, 288, 0, stream>>>(xh, srcA, et, time_W, time_b, msgs);
    run_gemm(msgs, wih16, gru_bih, gi, nullptr, NEDGES, 768, 288, 768, 0, 0, stream);

    // 5. sequential GRU per dst-chain (parallel across chains)
    gru_chains_kernel<<<NEDGES, 256, 0, stream>>>(keys, starts, counter, gi, gru_Whh, gru_bhh, mem);

    // 6. emb0 = [relu(x@featW^T+b) , mem]  (f16, ld=512)
    run_gemm(xh, featw16, feat_b, nullptr, emb0, NNODES, 256, 256, 512, 0, 1, stream);
    mem_to_emb_kernel<<<NNODES, 256, 0, stream>>>(mem, emb0);

    // -------- TransformerConv layers --------
    struct Layer {
        const _Float16 *in, *qW, *kW, *vW, *sW;
        const float *qb, *kb, *vb, *sb;
        int K; _Float16* out;
    } layers[2] = {
        { emb0, q0w16, k0w16, v0w16, s0w16, c0_qb, c0_kb, c0_vb, c0_sb, 512, emb1 },
        { emb1, q1w16, k1w16, v1w16, s1w16, c1_qb, c1_kb, c1_vb, c1_sb, 256, emb2 },
    };

    for (int l = 0; l < 2; ++l) {
        Layer& L = layers[l];
        // skip path over all N nodes (WMMA, f32 out)
        run_gemm(L.in, L.sW, L.sb, skip, nullptr, NNODES, 256, L.K, 256, 0, 0, stream);
        // gather only edge-touched rows, then q/k/v GEMMs on 8192 rows (6x FLOP cut vs reference)
        gather_rows_kernel<<<NEDGES, 256, 0, stream>>>(L.in, dstA, Xgd, L.K);
        gather_rows_kernel<<<NEDGES, 256, 0, stream>>>(L.in, srcA, Xgs, L.K);
        run_gemm(Xgd, L.qW, L.qb, qe, nullptr, NEDGES, 512, L.K, 512, 0, 0, stream);
        run_gemm(Xgs, L.kW, L.kb, ke, nullptr, NEDGES, 512, L.K, 512, 0, 0, stream);
        run_gemm(Xgs, L.vW, L.vb, ve, nullptr, NEDGES, 512, L.K, 512, 0, 0, stream);
        // segment softmax over incoming edges
        run_zero(menc,  (long)NNODES * 2, stream);
        run_zero(denom, (long)NNODES * 2, stream);
        run_zero(aggm,  (long)NNODES * 256, stream);
        attn_alpha_kernel<<<(NEDGES * 2 * 32 + 255) / 256, 256, 0, stream>>>(qe, ke, dstA, alpha, menc);
        attn_exp_kernel<<<(NEDGES * 2 + 255) / 256, 256, 0, stream>>>(alpha, dstA, menc, exv, denom);
        attn_agg_kernel<<<NEDGES, 256, 0, stream>>>(exv, denom, ve, dstA, aggm);
        // out = relu(head-mean(agg) + skip)
        combine_kernel<<<NNODES, 256, 0, stream>>>(aggm, skip, L.out);
    }

    // classifier: d_out = emb2 @ clsW^T + b  (f32 out)
    run_gemm(emb2, clsw16, cls_b, (float*)d_out, nullptr, NNODES, 64, 256, 64, 0, 0, stream);
}
